// MultiLayerHeteroGraphConv_77077483094301
// MI455X (gfx1250) — matrix operand order repaired and checked by reading, not compiled, not run on hardware
//
#include <hip/hip_runtime.h>

// Problem constants (match reference)
#define NU_   100000
#define NI_   50000
#define DD    128
#define EE    200000
#define CHUNK 65536   // edge-chunk rows for last-layer review MLP

typedef __attribute__((ext_vector_type(16))) _Float16 v16h;
typedef __attribute__((ext_vector_type(8)))  float    v8f;
typedef __attribute__((ext_vector_type(4)))  float    v4f;

__device__ __forceinline__ float gelu_exact(float x) {
    return 0.5f * x * (1.0f + erff(x * 0.70710678118654752440f));
}
__device__ __forceinline__ float sigmoid_(float x) {
    return 1.0f / (1.0f + __expf(-x));
}
__device__ __forceinline__ float wave_sum32(float v) {
#pragma unroll
    for (int off = 16; off > 0; off >>= 1) v += __shfl_xor(v, off, 32);
    return v;
}
__device__ __forceinline__ void atomAddF(float* p, float v) {
    __hip_atomic_fetch_add(p, v, __ATOMIC_RELAXED, __HIP_MEMORY_SCOPE_AGENT);
}

// ---------------------------------------------------------------------------
// Y[M,128] = rowScale ⊙ act(A) @ W^T (+ bias) ; optional post-GELU / accumulate
// W is [128,128] row-major; Y[m,n] = sum_k A[m,k] * W[n,k].
// One wave32 = one 16x128 output tile (8 N-tiles x 4 K-steps = 32 WMMA).
// All 8 B fragments of a K-step are batch-loaded before the WMMA chain so the
// 8 independent v_wmma issue back-to-back and the next K-step's ds_loads
// overlap with the matrix pipe instead of stalling on s_wait_dscnt per WMMA.
// ---------------------------------------------------------------------------
template <int PRE_GELU, int POST_GELU, int ACCUM>
__global__ __launch_bounds__(256) void gemm128_wmma(
    const float* __restrict__ A, const float* __restrict__ rowScale,
    const float* __restrict__ W, const float* __restrict__ bias,
    float* __restrict__ Y, int M)
{
    __shared__ _Float16 Wl[128 * 128];  // 32 KB: W staged as f16
    for (int i = threadIdx.x; i < (128 * 128) / 4; i += 256) {
        v4f w = ((const v4f*)W)[i];
        int b = i * 4;
        Wl[b + 0] = (_Float16)w[0];
        Wl[b + 1] = (_Float16)w[1];
        Wl[b + 2] = (_Float16)w[2];
        Wl[b + 3] = (_Float16)w[3];
    }
    __syncthreads();

    const int lane  = threadIdx.x & 31;
    const int wave  = threadIdx.x >> 5;
    const int khalf = lane >> 4;    // 0: lanes 0-15, 1: lanes 16-31
    const int mLane = lane & 15;
    const int mBase = (blockIdx.x * 8 + wave) * 16;

    const int   mRow = mBase + mLane;
    const int   mC   = (mRow < M) ? mRow : (M - 1);   // clamp; stores are masked
    const float* Arow = A + (size_t)mC * 128;
    const float  rs   = rowScale ? rowScale[mC] : 1.0f;

    // LDS base for this lane's B-fragment column within an N-tile:
    // row n = nt*16 + mLane, cols kb..kb+15 with kb = kt*32 + khalf*16.
    const _Float16* WlLane = &Wl[mLane * 128 + khalf * 16];

    v8f acc[8] = {};

#pragma unroll
    for (int kt = 0; kt < 4; ++kt) {
        // A fragment (contiguous k order, same as B): halfs 0..7  -> K = kt*32 + khalf*16 + j
        //                                     halfs 8..15 -> K = kt*32 + khalf*16 + 8 + j
        v16h a;
        const int k0 = kt * 32 + khalf * 16;
#pragma unroll
        for (int j = 0; j < 8; ++j) {
            float x0 = Arow[k0 + j];
            float x1 = Arow[k0 + 8 + j];
            if (PRE_GELU) { x0 = gelu_exact(x0); x1 = gelu_exact(x1); }
            a[j]     = (_Float16)(x0 * rs);
            a[8 + j] = (_Float16)(x1 * rs);
        }

        // Batch-load all 8 B fragments of this K-step (16-bit B 32x16 layout).
        v16h barr[8];
#pragma unroll
        for (int nt = 0; nt < 8; ++nt) {
            barr[nt] = *(const v16h*)(WlLane + nt * (16 * 128) + kt * 32);
        }
        // 8 independent WMMAs: no cross-RAW, issue back-to-back on the matrix pipe.
#pragma unroll
        for (int nt = 0; nt < 8; ++nt) {
            acc[nt] = __builtin_amdgcn_wmma_f32_16x16x32_f16(
                false, a, false, barr[nt], (short)0, acc[nt], false, false);
        }
    }

    // D layout: element v -> row = mBase + khalf*8 + v, col = nt*16 + mLane
#pragma unroll
    for (int nt = 0; nt < 8; ++nt) {
        const int col = nt * 16 + mLane;
#pragma unroll
        for (int v = 0; v < 8; ++v) {
            const int row = mBase + khalf * 8 + v;
            if (row < M) {
                float val = acc[nt][v];
                if (bias) val += bias[col];
                if (POST_GELU) val = gelu_exact(val);
                float* yp = Y + (size_t)row * 128 + col;
                if (ACCUM) *yp += val; else *yp = val;
            }
        }
    }
}

// ---------------------------------------------------------------------------
// Non-last-layer edge message: m = h[src] * sigmoid(<rf128, pw>) * c_src[src]
// scatter-added into agg[dst]. One wave32 per edge, 4 floats per lane.
// ---------------------------------------------------------------------------
__global__ __launch_bounds__(256) void edge_msg_kernel(
    const int* __restrict__ src, const int* __restrict__ dst,
    const int* __restrict__ rid, int nE,
    const float* __restrict__ h, const float* __restrict__ review,
    const float* __restrict__ pw, const float* __restrict__ c_src,
    float* __restrict__ agg)
{
    const int lane = threadIdx.x & 31;
    const int e = blockIdx.x * 8 + (threadIdx.x >> 5);
    if (e >= nE) return;
    const int rv = rid[e];
    v4f x = ((const v4f*)(review + (size_t)rv * 128))[lane];
    v4f p = ((const v4f*)pw)[lane];
    float d = wave_sum32(x[0]*p[0] + x[1]*p[1] + x[2]*p[2] + x[3]*p[3]);
    const int   s     = src[e];
    const float scale = sigmoid_(d) * c_src[s];
    v4f hv = ((const v4f*)(h + (size_t)s * 128))[lane];
    float* out = agg + (size_t)dst[e] * 128 + lane * 4;
    atomAddF(out + 0, hv[0] * scale);
    atomAddF(out + 1, hv[1] * scale);
    atomAddF(out + 2, hv[2] * scale);
    atomAddF(out + 3, hv[3] * scale);
}

// Last layer, stage 1: gather rf128 chunk + compute pa/ra gates per edge
__global__ __launch_bounds__(256) void edge_gather_kernel(
    const int* __restrict__ rid, int e0, int n,
    const float* __restrict__ review, const float* __restrict__ pw,
    const float* __restrict__ rsw,
    float* __restrict__ R, float* __restrict__ pa, float* __restrict__ ra)
{
    const int lane = threadIdx.x & 31;
    const int i = blockIdx.x * 8 + (threadIdx.x >> 5);
    if (i >= n) return;
    const int rv = rid[e0 + i];
    v4f x = ((const v4f*)(review + (size_t)rv * 128))[lane];
    ((v4f*)(R + (size_t)i * 128))[lane] = x;
    v4f p = ((const v4f*)pw)[lane];
    v4f q = ((const v4f*)rsw)[lane];
    float d1 = wave_sum32(x[0]*p[0] + x[1]*p[1] + x[2]*p[2] + x[3]*p[3]);
    float d2 = wave_sum32(x[0]*q[0] + x[1]*q[1] + x[2]*q[2] + x[3]*q[3]);
    if (lane == 0) { pa[i] = sigmoid_(d1); ra[i] = sigmoid_(d2); }
}

// Last layer, stage 3: m = (h[src]*pa + RF*ra) * c_src[src] -> agg[dst]
__global__ __launch_bounds__(256) void edge_combine_kernel(
    const int* __restrict__ src, const int* __restrict__ dst, int e0, int n,
    const float* __restrict__ h, const float* __restrict__ RF,
    const float* __restrict__ pa, const float* __restrict__ ra,
    const float* __restrict__ c_src, float* __restrict__ agg)
{
    const int lane = threadIdx.x & 31;
    const int i = blockIdx.x * 8 + (threadIdx.x >> 5);
    if (i >= n) return;
    const int e = e0 + i;
    const int s = src[e];
    const float pav = pa[i], rav = ra[i], cs = c_src[s];
    v4f hv = ((const v4f*)(h + (size_t)s * 128))[lane];
    v4f rf = ((const v4f*)(RF + (size_t)i * 128))[lane];
    float* out = agg + (size_t)dst[e] * 128 + lane * 4;
    atomAddF(out + 0, (hv[0] * pav + rf[0] * rav) * cs);
    atomAddF(out + 1, (hv[1] * pav + rf[1] * rav) * cs);
    atomAddF(out + 2, (hv[2] * pav + rf[2] * rav) * cs);
    atomAddF(out + 3, (hv[3] * pav + rf[3] * rav) * cs);
}

__global__ __launch_bounds__(256) void zero_f32(float* __restrict__ p, long long n)
{
    long long i = (long long)blockIdx.x * blockDim.x + threadIdx.x;
    const long long stride = (long long)gridDim.x * blockDim.x;
    for (; i < n; i += stride) p[i] = 0.0f;
}

// ---------------------------------------------------------------------------
extern "C" void kernel_launch(void* const* d_in, const int* in_sizes, int n_in,
                              void* d_out, int out_size, void* d_ws, size_t ws_size,
                              hipStream_t stream)
{
    (void)in_sizes; (void)n_in; (void)out_size; (void)ws_size;

    const float* user_emb    = (const float*)d_in[0];
    const float* item_emb    = (const float*)d_in[1];
    const float* review      = (const float*)d_in[2];
    const float* c_user      = (const float*)d_in[3];
    const float* c_item      = (const float*)d_in[4];
    const float* prob_w      = (const float*)d_in[5];   // [2][6][128]
    const float* lin_w       = (const float*)d_in[6];   // [2][6][128][128]
    const float* lin_b       = (const float*)d_in[7];   // [2][6][128]
    const float* emb_w       = (const float*)d_in[8];   // [6][128][128]
    const float* emb_b       = (const float*)d_in[9];   // [6][128]
    const float* rev_score_w = (const float*)d_in[10];  // [6][128]
    const float* rev_w1      = (const float*)d_in[11];  // [6][128][128]
    const float* rev_w2      = (const float*)d_in[12];
    const float* rev_w3      = (const float*)d_in[13];
    const float* ufc_w       = (const float*)d_in[14];
    const float* ufc_b       = (const float*)d_in[15];
    const float* ifc_w       = (const float*)d_in[16];
    const float* ifc_b       = (const float*)d_in[17];
    const int*   src_u2i     = (const int*)d_in[18];    // [2][3][E]
    const int*   dst_u2i     = (const int*)d_in[19];
    const int*   rid_u2i     = (const int*)d_in[20];
    const int*   src_i2u     = (const int*)d_in[21];
    const int*   dst_i2u     = (const int*)d_in[22];
    const int*   rid_i2u     = (const int*)d_in[23];
    // d_in[24]/d_in[25] (in_u/in_v) are identity arange permutations.

    float* out = (float*)d_out;

    // Workspace layout (floats). Layer-0 h buffers alias the layer-1 accumulators.
    float* ws = (float*)d_ws;
    const size_t NUD = (size_t)NU_ * DD, NID = (size_t)NI_ * DD;
    float* uf[3] = { ws, ws + NUD, ws + 2 * NUD };               // layer-0 user outs
    float* vf[3] = { ws + 3*NUD, ws + 3*NUD + NID, ws + 3*NUD + 2*NID };
    float* u_acc = ws + 3 * NUD + 3 * NID;  // layer-1 sum over r (also hU in layer 0)
    float* i_acc = u_acc + NUD;             // (also hI in layer 0)
    float* aggU  = i_acc + NID;
    float* aggI  = aggU + NUD;
    float* eA    = aggI + NID;              // CHUNK x 128
    float* eB    = eA + (size_t)CHUNK * DD; // CHUNK x 128
    float* paB   = eB + (size_t)CHUNK * DD;
    float* raB   = paB + CHUNK;
    float* hU = u_acc;  // layer-0 only
    float* hI = i_acc;  // layer-0 only

    auto zero = [&](float* p, size_t n) {
        zero_f32<<<dim3(2048), dim3(256), 0, stream>>>(p, (long long)n);
    };
    auto gemm = [&](const float* A, const float* sc, const float* W, const float* b,
                    float* Y, int M, int pre, int post, int accum) {
        dim3 g((unsigned)((M + 127) / 128)), blk(256);
        if (pre)        gemm128_wmma<1,0,0><<<g, blk, 0, stream>>>(A, sc, W, b, Y, M);
        else if (post)  gemm128_wmma<0,1,0><<<g, blk, 0, stream>>>(A, sc, W, b, Y, M);
        else if (accum) gemm128_wmma<0,0,1><<<g, blk, 0, stream>>>(A, sc, W, b, Y, M);
        else            gemm128_wmma<0,0,0><<<g, blk, 0, stream>>>(A, sc, W, b, Y, M);
    };
    auto egrid = [](int n) { return dim3((unsigned)((n + 7) / 8)); };

    // ----------------------- Layer 0 (first=True) -----------------------
    for (int r = 0; r < 3; ++r) {
        // user -> item  (k = r)
        int k = r;
        gemm(user_emb, nullptr, emb_w + (size_t)k*DD*DD, emb_b + k*DD, hU, NU_, 0, 0, 0);
        zero(aggI, NID);
        edge_msg_kernel<<<egrid(EE), 256, 0, stream>>>(
            src_u2i + (size_t)r*EE, dst_u2i + (size_t)r*EE, rid_u2i + (size_t)r*EE, EE,
            hU, review, prob_w + (size_t)k*DD, c_user, aggI);
        gemm(aggI, c_item, lin_w + (size_t)k*DD*DD, lin_b + k*DD, vf[r], NI_, 0, 0, 0);

        // item -> user  (k = 3 + r)
        k = 3 + r;
        gemm(item_emb, nullptr, emb_w + (size_t)k*DD*DD, emb_b + k*DD, hI, NI_, 0, 0, 0);
        zero(aggU, NUD);
        edge_msg_kernel<<<egrid(EE), 256, 0, stream>>>(
            src_i2u + (size_t)r*EE, dst_i2u + (size_t)r*EE, rid_i2u + (size_t)r*EE, EE,
            hI, review, prob_w + (size_t)k*DD, c_item, aggU);
        gemm(aggU, c_user, lin_w + (size_t)k*DD*DD, lin_b + k*DD, uf[r], NU_, 0, 0, 0);
    }

    // ----------------------- Layer 1 (last=True) -----------------------
    zero(u_acc, NUD);
    zero(i_acc, NID);
    for (int r = 0; r < 3; ++r) {
        // user -> item  (k = r, l = 1), h = uf[r]
        {
            const int k = r;
            const int* sp = src_u2i + (size_t)(3 + r) * EE;
            const int* dp = dst_u2i + (size_t)(3 + r) * EE;
            const int* rp = rid_u2i + (size_t)(3 + r) * EE;
            zero(aggI, NID);
            for (int e0 = 0; e0 < EE; e0 += CHUNK) {
                const int n = (EE - e0 < CHUNK) ? (EE - e0) : CHUNK;
                edge_gather_kernel<<<egrid(n), 256, 0, stream>>>(
                    rp, e0, n, review, prob_w + (size_t)(6 + k)*DD,
                    rev_score_w + (size_t)k*DD, eA, paB, raB);
                gemm(eA, nullptr, rev_w1 + (size_t)k*DD*DD, nullptr, eB, n, 0, 1, 0);
                gemm(eB, nullptr, rev_w2 + (size_t)k*DD*DD, nullptr, eA, n, 0, 1, 0);
                gemm(eA, nullptr, rev_w3 + (size_t)k*DD*DD, nullptr, eB, n, 0, 0, 0);
                edge_combine_kernel<<<egrid(n), 256, 0, stream>>>(
                    sp, dp, e0, n, uf[r], eB, paB, raB, c_user, aggI);
            }
            gemm(aggI, c_item, lin_w + (size_t)(6 + k)*DD*DD, lin_b + (size_t)(6 + k)*DD,
                 i_acc, NI_, 0, 0, 1);
        }
        // item -> user  (k = 3 + r, l = 1), h = vf[r]
        {
            const int k = 3 + r;
            const int* sp = src_i2u + (size_t)(3 + r) * EE;
            const int* dp = dst_i2u + (size_t)(3 + r) * EE;
            const int* rp = rid_i2u + (size_t)(3 + r) * EE;
            zero(aggU, NUD);
            for (int e0 = 0; e0 < EE; e0 += CHUNK) {
                const int n = (EE - e0 < CHUNK) ? (EE - e0) : CHUNK;
                edge_gather_kernel<<<egrid(n), 256, 0, stream>>>(
                    rp, e0, n, review, prob_w + (size_t)(6 + k)*DD,
                    rev_score_w + (size_t)k*DD, eA, paB, raB);
                gemm(eA, nullptr, rev_w1 + (size_t)k*DD*DD, nullptr, eB, n, 0, 1, 0);
                gemm(eB, nullptr, rev_w2 + (size_t)k*DD*DD, nullptr, eA, n, 0, 1, 0);
                gemm(eA, nullptr, rev_w3 + (size_t)k*DD*DD, nullptr, eB, n, 0, 0, 0);
                edge_combine_kernel<<<egrid(n), 256, 0, stream>>>(
                    sp, dp, e0, n, vf[r], eB, paB, raB, c_item, aggU);
            }
            gemm(aggU, c_user, lin_w + (size_t)(6 + k)*DD*DD, lin_b + (size_t)(6 + k)*DD,
                 u_acc, NU_, 0, 0, 1);
        }
    }

    // ----------------------- Final FCs: out = gelu(acc) @ W^T + b ------
    gemm(u_acc, nullptr, ufc_w, ufc_b, out, NU_, 1, 0, 0);
    gemm(i_acc, nullptr, ifc_w, ifc_b, out + NUD, NI_, 1, 0, 0);
}
